// HierNet_55705725829422
// MI455X (gfx1250) — matrix operand order repaired and checked by reading, not compiled, not run on hardware
//
#include <hip/hip_runtime.h>

// ---------------------------------------------------------------------------
// Problem constants (match reference)
// ---------------------------------------------------------------------------
#define NN 50000
#define EE 800000
#define GG 64
#define AVG_LOG 2.8332133440562162f

typedef __bf16 bf16;
typedef __attribute__((ext_vector_type(16))) __bf16 v16bf;
typedef __attribute__((ext_vector_type(8)))  __bf16 v8bf;
typedef __attribute__((ext_vector_type(8)))  float  v8f;
typedef __attribute__((ext_vector_type(4))) unsigned int u32x4;
typedef __attribute__((ext_vector_type(4))) int i32x4;
typedef __attribute__((ext_vector_type(8))) int i32x8;

// Order-preserving float <-> uint key (for atomic min/max over floats)
__device__ __forceinline__ unsigned fkey(float f) {
  unsigned u = __float_as_uint(f);
  return (u & 0x80000000u) ? ~u : (u | 0x80000000u);
}
__device__ __forceinline__ float funkey(unsigned k) {
  unsigned u = (k & 0x80000000u) ? (k & 0x7fffffffu) : ~k;
  return __uint_as_float(u);
}

// Load a 16x32 bf16 A-fragment slice for this lane from an LDS row.
// ISA layout: lane<16 -> K {0..7,16..23}, lane>=16 -> K {8..15,24..31}.
__device__ __forceinline__ v16bf load_a_frag(const bf16* base, int hs) {
  v8bf lo = *(const v8bf*)(base + hs * 8);
  v8bf hi = *(const v8bf*)(base + 16 + hs * 8);
  v16bf a;
#pragma unroll
  for (int i = 0; i < 8; i++) { a[i] = lo[i]; a[8 + i] = hi[i]; }
  return a;
}

__device__ __forceinline__ v16bf load_b_frag(const bf16* p) {
  return *(const v16bf*)p; // 32 contiguous bytes per lane
}

// ---------------------------------------------------------------------------
// Tensor Data Mover: gather 16 rows of F bf16 elements from a [NN, F] bf16
// tensor into LDS, padding 2F elements after each row so rows land at stride
// 3F inside the WMMA A staging tile.  D# per CDNA5 ISA ch.8 (gather mode,
// 16-bit indices, data_size=2B).  6-arg builtin form (clang-23 toolchain).
// ---------------------------------------------------------------------------
template <int F>
__device__ __forceinline__ void tdm_gather_rows(unsigned lds_byte_addr,
                                                unsigned long long gaddr,
                                                i32x4 g2, i32x4 g3) {
  constexpr unsigned padI = (F == 64) ? 4u : 3u;   // row = 32 / 16 dwords
  constexpr unsigned padA = (F == 64) ? 63u : 31u; // pad = 64 / 32 dwords (2F elems)
  u32x4 g0;
  g0[0] = 1u | (1u << 31);                          // count=1, gather_mode=1, idx16
  g0[1] = lds_byte_addr;
  g0[2] = (unsigned)(gaddr & 0xffffffffull);
  g0[3] = (unsigned)((gaddr >> 32) & 0x1ffffffull) | (2u << 30); // addr[56:32], type=2
  unsigned long long q0 = (1ull << 16)                       // data_size = 2 bytes
                        | (1ull << 20)                       // pad_enable
                        | ((unsigned long long)padI << 22)
                        | ((unsigned long long)padA << 25)
                        | ((unsigned long long)(F & 0xffff) << 48);  // tensor_dim0 lo16
  unsigned long long q1 = ((unsigned long long)NN << 16)             // tensor_dim1
                        | ((unsigned long long)(F & 0xffff) << 48);  // tile_dim0
  unsigned long long q2 = 16ull                                      // tile_dim1 = #indices
                        | ((unsigned long long)(unsigned)F << 32);   // dim0_stride lo32
  unsigned long long q3 = 0ull;                                      // stride hi, dim1_stride
  i32x8 g1;
  g1[0] = (int)q0; g1[1] = (int)(q0 >> 32);
  g1[2] = (int)q1; g1[3] = (int)(q1 >> 32);
  g1[4] = (int)q2; g1[5] = (int)(q2 >> 32);
  g1[6] = (int)q3; g1[7] = (int)(q3 >> 32);
  i32x8 gz = {};                                     // unused trailing group
  __builtin_amdgcn_tensor_load_to_lds(g0, g1, g2, g3, gz, 0);
}

// Pack 8 int32 node indices (wave-uniform addresses) into 4 dwords of
// 16-bit gather indices, forced into SGPRs.
__device__ __forceinline__ i32x4 pack_idx(const int* __restrict__ p) {
  i32x4 r;
#pragma unroll
  for (int j = 0; j < 4; j++) {
    int lo = p[2 * j], hi = p[2 * j + 1];
    r[j] = __builtin_amdgcn_readfirstlane((lo & 0xffff) | (hi << 16));
  }
  return r;
}

// ---------------------------------------------------------------------------
// Pack a row-major weight matrix W[T][K][Ncol] (f32) into bf16 B-fragment
// layout: out[(((t*KC+kc)*NC+nc)*32+lane)*16+k2] = W[t][kc*32+(lane>>4)*8+(k2&7)+(k2>>3)*16]
//                                                    [nc*16+(lane&15)]
// ---------------------------------------------------------------------------
__global__ void pack_b_kernel(const float* __restrict__ W, bf16* __restrict__ out,
                              int T, int K, int Ncol) {
  int total = T * K * Ncol;
  int NC = Ncol >> 4, KC = K >> 5;
  for (int i = blockIdx.x * blockDim.x + threadIdx.x; i < total;
       i += gridDim.x * blockDim.x) {
    int k2   = i & 15;
    int lane = (i >> 4) & 31;
    int rest = i >> 9;
    int nc = rest % NC; rest /= NC;
    int kc = rest % KC;
    int t  = rest / KC;
    int k = kc * 32 + (lane >> 4) * 8 + (k2 & 7) + (k2 >> 3) * 16;
    int n = nc * 16 + (lane & 15);
    out[i] = (bf16)W[(t * K + k) * Ncol + n];
  }
}

__global__ void fill_kernel(unsigned* __restrict__ p, unsigned v, long n) {
  for (long i = blockIdx.x * (long)blockDim.x + threadIdx.x; i < n;
       i += (long)gridDim.x * blockDim.x)
    p[i] = v;
}

__global__ void cvt_bf16_kernel(const float* __restrict__ in, bf16* __restrict__ out,
                                long n) {
  for (long i = blockIdx.x * (long)blockDim.x + threadIdx.x; i < n;
       i += (long)gridDim.x * blockDim.x)
    out[i] = (bf16)in[i];
}

__global__ void degree_kernel(const int* __restrict__ dst, float* __restrict__ cnt) {
  int i = blockIdx.x * blockDim.x + threadIdx.x;
  if (i < EE) atomicAdd(&cnt[dst[i]], 1.0f);
}

// ---------------------------------------------------------------------------
// Fused edge kernel.  2 waves/block; each wave owns MT=4 16-edge tiles.
// TDM gathers the bf16 feature rows of dst/src straight into the A staging
// tiles in LDS (async, TENSORcnt); the wave fills the edge-encoder columns,
// then runs the message GEMM with one B-fragment load feeding 4 WMMAs, and
// scatters sum/sum^2/min/max via L2 atomics.
// ---------------------------------------------------------------------------
template <int F>
__global__ __launch_bounds__(64) void edge_kernel(
    const bf16*  __restrict__ xb,     // [N,F] bf16 node features
    const float* __restrict__ eattr,  // [E,2]
    const float* __restrict__ We,     // [2,F]
    const float* __restrict__ be,     // [F]
    const bf16*  __restrict__ WpreP,  // packed [T,3F,F]
    const float* __restrict__ bpre,   // [T,F]
    const int* __restrict__ src, const int* __restrict__ dst,
    float* __restrict__ sumB, float* __restrict__ sqB,
    unsigned* __restrict__ minB, unsigned* __restrict__ maxB) {
  constexpr int K3F = 3 * F;
  constexpr int KC = K3F / 32;
  constexpr int NC = F / 16;
  constexpr int TF = 2 * F;
  constexpr int MT = 4;                          // M tiles per wave
  __shared__ __align__(16) bf16 sA[2][MT * 16 * K3F];

  const int wave = threadIdx.x >> 5;
  const int lane = threadIdx.x & 31;
  const int e0 = (blockIdx.x * 2 + wave) * (MT * 16);

  bf16* A = sA[wave];
  unsigned ldsBase =
      (unsigned)__builtin_amdgcn_readfirstlane((int)(unsigned)(size_t)(void*)A);
  const unsigned long long gx = (unsigned long long)(size_t)xb;

  // Issue 2 TDM row-gathers per tile: dst rows -> cols [0,F), src -> [F,2F)
#pragma unroll
  for (int mt = 0; mt < MT; mt++) {
    int te0 = e0 + mt * 16;
    unsigned tileLds = ldsBase + (unsigned)(mt * 16 * K3F * 2);
    tdm_gather_rows<F>(tileLds, gx, pack_idx(dst + te0), pack_idx(dst + te0 + 8));
    tdm_gather_rows<F>(tileLds + 2 * F, gx, pack_idx(src + te0),
                       pack_idx(src + te0 + 8));
  }

  // Edge-encoder columns [2F,3F) while TDM streams in
  for (int r = 0; r < MT * 16; r++) {
    int e = e0 + r;
    float ea0 = eattr[2 * e], ea1 = eattr[2 * e + 1];
    for (int c = lane; c < F; c += 32)
      A[r * K3F + 2 * F + c] = (bf16)(ea0 * We[c] + ea1 * We[F + c] + be[c]);
  }
  __builtin_amdgcn_s_wait_tensorcnt(0);
  __syncthreads();

  const int row = lane & 15, hs = lane >> 4;
  int myDst[MT][8];
#pragma unroll
  for (int mt = 0; mt < MT; mt++)
#pragma unroll
    for (int v = 0; v < 8; v++) myDst[mt][v] = dst[e0 + mt * 16 + hs * 8 + v];

  for (int t = 0; t < 2; t++) {
#pragma unroll
    for (int nc = 0; nc < NC; nc++) {
      v8f acc[MT];
#pragma unroll
      for (int mt = 0; mt < MT; mt++) acc[mt] = (v8f){};
#pragma unroll
      for (int kc = 0; kc < KC; kc++) {
        v16bf b = load_b_frag(WpreP + (((t * KC + kc) * NC + nc) * 32 + lane) * 16);
#pragma unroll
        for (int mt = 0; mt < MT; mt++) {
          v16bf a = load_a_frag(&A[(mt * 16 + row) * K3F + kc * 32], hs);
          acc[mt] = __builtin_amdgcn_wmma_f32_16x16x32_bf16(
              false, a, false, b, (short)0, acc[mt], false, false);
        }
      }
      int n = nc * 16 + row;
      float bias = bpre[t * F + n];
#pragma unroll
      for (int mt = 0; mt < MT; mt++) {
#pragma unroll
        for (int v = 0; v < 8; v++) {
          float val = acc[mt][v] + bias;   // m[e, t, n]
          int col = myDst[mt][v] * TF + t * F + n;
          atomicAdd(&sumB[col], val);
          atomicAdd(&sqB[col], val * val);
          unsigned k = fkey(val);
          atomicMax(&maxB[col], k);
          atomicMin(&minB[col], k);
        }
      }
    }
  }
}

// ---------------------------------------------------------------------------
// Node kernel: stats -> 13F feature staging -> Wpost WMMA (per tower)
// -> Wlin WMMA -> ReLU -> hout[N,64] (f32) + houtb[N,64] (bf16 for next
// layer's TDM gather).  One wave per 16-node tile.
// ---------------------------------------------------------------------------
template <int F>
__global__ __launch_bounds__(32) void node_kernel(
    const float* __restrict__ xin,    // [N,F] layer input (x_t feature)
    const float* __restrict__ sumB, const float* __restrict__ sqB,
    const unsigned* __restrict__ minB, const unsigned* __restrict__ maxB,
    const float* __restrict__ cnt,
    const bf16*  __restrict__ WpostP, // packed [T,13F,32]
    const float* __restrict__ bpost,  // [T,32]
    const bf16*  __restrict__ WlinP,  // packed [64,64]
    const float* __restrict__ blin,   // [64]
    float* __restrict__ hout,         // [N,64]
    bf16* __restrict__ houtb) {       // [N,64]
  constexpr int K13 = 13 * F;
  constexpr int KC = K13 / 32;
  constexpr int TF = 2 * F;
  __shared__ __align__(16) bf16 sF[16 * K13];
  __shared__ __align__(16) bf16 sO[16 * 64];

  const int lane = threadIdx.x;
  const int n0 = blockIdx.x * 16;
  const int row = lane & 15, hs = lane >> 4;

  for (int t = 0; t < 2; t++) {
    // Build feature rows: [x | mean,mn,mx,std | *amp | *att]
    for (int idx = lane; idx < 16 * F; idx += 32) {
      int i = idx / F, j = idx % F;
      int node = n0 + i;
      float c = cnt[node];
      float d = fmaxf(c, 1.0f);
      int col = node * TF + t * F + j;
      float mean = sumB[col] / d;
      float msq = sqB[col] / d;
      float stdv = sqrtf(fmaxf(msq - mean * mean, 0.0f) + 1e-5f);
      bool has = c > 0.0f;
      float mn = has ? funkey(minB[col]) : 0.0f;
      float mx = has ? funkey(maxB[col]) : 0.0f;
      float logd = logf(d + 1.0f);
      float amp = logd / AVG_LOG;
      float att = AVG_LOG / logd;
      bf16* r = &sF[i * K13];
      r[j]          = (bf16)xin[node * F + j];
      r[F + j]      = (bf16)mean;
      r[2 * F + j]  = (bf16)mn;
      r[3 * F + j]  = (bf16)mx;
      r[4 * F + j]  = (bf16)stdv;
      r[5 * F + j]  = (bf16)(mean * amp);
      r[6 * F + j]  = (bf16)(mn * amp);
      r[7 * F + j]  = (bf16)(mx * amp);
      r[8 * F + j]  = (bf16)(stdv * amp);
      r[9 * F + j]  = (bf16)(mean * att);
      r[10 * F + j] = (bf16)(mn * att);
      r[11 * F + j] = (bf16)(mx * att);
      r[12 * F + j] = (bf16)(stdv * att);
    }
    __syncthreads();
    // Wpost[t]: [16 x 13F] @ [13F x 32]
#pragma unroll
    for (int nc = 0; nc < 2; nc++) {
      v8f acc = {};
#pragma unroll
      for (int kc = 0; kc < KC; kc++) {
        v16bf a = load_a_frag(&sF[row * K13 + kc * 32], hs);
        v16bf b = load_b_frag(WpostP + (((t * KC + kc) * 2 + nc) * 32 + lane) * 16);
        acc = __builtin_amdgcn_wmma_f32_16x16x32_bf16(false, a, false, b,
                                                      (short)0, acc, false, false);
      }
      int n = nc * 16 + row;
      float bias = bpost[t * 32 + n];
#pragma unroll
      for (int v = 0; v < 8; v++)
        sO[(hs * 8 + v) * 64 + t * 32 + n] = (bf16)(acc[v] + bias);
    }
    __syncthreads();
  }
  // Wlin: [16 x 64] @ [64 x 64], + blin, ReLU
#pragma unroll
  for (int nc = 0; nc < 4; nc++) {
    v8f acc = {};
#pragma unroll
    for (int kc = 0; kc < 2; kc++) {
      v16bf a = load_a_frag(&sO[row * 64 + kc * 32], hs);
      v16bf b = load_b_frag(WlinP + ((kc * 4 + nc) * 32 + lane) * 16);
      acc = __builtin_amdgcn_wmma_f32_16x16x32_bf16(false, a, false, b,
                                                    (short)0, acc, false, false);
    }
    int n = nc * 16 + row;
    float bias = blin[n];
#pragma unroll
    for (int v = 0; v < 8; v++) {
      float val = fmaxf(acc[v] + bias, 0.0f);
      int node = n0 + hs * 8 + v;
      hout[node * 64 + n] = val;
      houtb[node * 64 + n] = (bf16)val;
    }
  }
}

// ---------------------------------------------------------------------------
// global_add_pool over batch
// ---------------------------------------------------------------------------
__global__ void pool_kernel(const float* __restrict__ h, const int* __restrict__ batch,
                            float* __restrict__ gbuf) {
  int i = blockIdx.x * blockDim.x + threadIdx.x;
  if (i < NN * 64) {
    int node = i >> 6, j = i & 63;
    atomicAdd(&gbuf[batch[node] * 64 + j], h[i]);
  }
}

// ---------------------------------------------------------------------------
// Final 3-layer MLP over G=64 graphs (one thread per graph; tiny)
// ---------------------------------------------------------------------------
__global__ __launch_bounds__(64) void mlp_kernel(
    const float* __restrict__ gbuf, const float* __restrict__ hls,
    const float* __restrict__ W1, const float* __restrict__ b1,
    const float* __restrict__ W2, const float* __restrict__ b2,
    const float* __restrict__ W3, const float* __restrict__ b3,
    float* __restrict__ out) {
  int g = threadIdx.x;
  if (g >= GG) return;
  float in[96];
#pragma unroll
  for (int j = 0; j < 64; j++) in[j] = gbuf[g * 64 + j];
#pragma unroll
  for (int j = 0; j < 32; j++) in[64 + j] = hls[g * 32 + j];
  float h1[64];
  for (int o = 0; o < 64; o++) {
    float acc = b1[o];
#pragma unroll
    for (int k = 0; k < 96; k++) acc += in[k] * W1[k * 64 + o];
    h1[o] = fmaxf(acc, 0.0f);
  }
  float h2[64];
  for (int o = 0; o < 64; o++) {
    float acc = b2[o];
#pragma unroll
    for (int k = 0; k < 64; k++) acc += h1[k] * W2[k * 64 + o];
    h2[o] = fmaxf(acc, 0.0f);
  }
  float acc = b3[0];
#pragma unroll
  for (int k = 0; k < 64; k++) acc += h2[k] * W3[k];
  out[g] = acc;
}

// ---------------------------------------------------------------------------
// Launch
// ---------------------------------------------------------------------------
extern "C" void kernel_launch(void* const* d_in, const int* in_sizes, int n_in,
                              void* d_out, int out_size, void* d_ws, size_t ws_size,
                              hipStream_t stream) {
  (void)in_sizes; (void)n_in; (void)out_size; (void)ws_size;
  const float* x     = (const float*)d_in[0];
  const float* eattr = (const float*)d_in[1];
  const float* hls   = (const float*)d_in[2];
  const int*   eidx  = (const int*)d_in[3];
  const int*   batch = (const int*)d_in[4];
  const float* We0 = (const float*)d_in[5],  *be0 = (const float*)d_in[6];
  const float* Wpre0 = (const float*)d_in[7], *bpre0 = (const float*)d_in[8];
  const float* Wpost0 = (const float*)d_in[9], *bpost0 = (const float*)d_in[10];
  const float* Wlin0 = (const float*)d_in[11], *blin0 = (const float*)d_in[12];
  const float* We1 = (const float*)d_in[13], *be1 = (const float*)d_in[14];
  const float* Wpre1 = (const float*)d_in[15], *bpre1 = (const float*)d_in[16];
  const float* Wpost1 = (const float*)d_in[17], *bpost1 = (const float*)d_in[18];
  const float* Wlin1 = (const float*)d_in[19], *blin1 = (const float*)d_in[20];
  const float* W1 = (const float*)d_in[21], *b1 = (const float*)d_in[22];
  const float* W2 = (const float*)d_in[23], *b2 = (const float*)d_in[24];
  const float* W3 = (const float*)d_in[25], *b3 = (const float*)d_in[26];
  const int* src = eidx;
  const int* dst = eidx + EE;

  char* w = (char*)d_ws;
  size_t off = 0;
  float* h0 = (float*)(w + off);         off += (size_t)NN * 64 * 4;
  float* h1 = (float*)(w + off);         off += (size_t)NN * 64 * 4;
  float* sumB = (float*)(w + off);       off += (size_t)NN * 128 * 4;
  float* sqB = (float*)(w + off);        off += (size_t)NN * 128 * 4;
  unsigned* minB = (unsigned*)(w + off); off += (size_t)NN * 128 * 4;
  unsigned* maxB = (unsigned*)(w + off); off += (size_t)NN * 128 * 4;
  float* cnt = (float*)(w + off);        off += (size_t)NN * 4;
  float* gbuf = (float*)(w + off);       off += (size_t)GG * 64 * 4;
  bf16* xb  = (bf16*)(w + off);          off += (size_t)NN * 32 * 2;
  bf16* h0b = (bf16*)(w + off);          off += (size_t)NN * 64 * 2;
  bf16* h1b = (bf16*)(w + off);          off += (size_t)NN * 64 * 2;
  bf16* WpreP0 = (bf16*)(w + off);       off += 2 * 96 * 32 * 2;
  bf16* WpreP1 = (bf16*)(w + off);       off += 2 * 192 * 64 * 2;
  bf16* WpostP0 = (bf16*)(w + off);      off += 2 * 416 * 32 * 2;
  bf16* WpostP1 = (bf16*)(w + off);      off += 2 * 832 * 32 * 2;
  bf16* WlinP0 = (bf16*)(w + off);       off += 64 * 64 * 2;
  bf16* WlinP1 = (bf16*)(w + off);       off += 64 * 64 * 2;

  // Pack weights into B-fragment layout (bf16); convert x to bf16
  pack_b_kernel<<<24, 256, 0, stream>>>(Wpre0, WpreP0, 2, 96, 32);
  pack_b_kernel<<<96, 256, 0, stream>>>(Wpre1, WpreP1, 2, 192, 64);
  pack_b_kernel<<<104, 256, 0, stream>>>(Wpost0, WpostP0, 2, 416, 32);
  pack_b_kernel<<<208, 256, 0, stream>>>(Wpost1, WpostP1, 2, 832, 32);
  pack_b_kernel<<<16, 256, 0, stream>>>(Wlin0, WlinP0, 1, 64, 64);
  pack_b_kernel<<<16, 256, 0, stream>>>(Wlin1, WlinP1, 1, 64, 64);
  cvt_bf16_kernel<<<1024, 256, 0, stream>>>(x, xb, (long)NN * 32);

  // Degree (shared by both layers) and pooled-accumulator init
  fill_kernel<<<256, 256, 0, stream>>>((unsigned*)cnt, 0u, NN);
  fill_kernel<<<64, 256, 0, stream>>>((unsigned*)gbuf, 0u, GG * 64);
  degree_kernel<<<EE / 256, 256, 0, stream>>>(dst, cnt);

  const long AGG = (long)NN * 128;
  // ---- Layer 0 (F=32) ----
  fill_kernel<<<2048, 256, 0, stream>>>((unsigned*)sumB, 0u, AGG);
  fill_kernel<<<2048, 256, 0, stream>>>((unsigned*)sqB, 0u, AGG);
  fill_kernel<<<2048, 256, 0, stream>>>(minB, 0xFFFFFFFFu, AGG);
  fill_kernel<<<2048, 256, 0, stream>>>(maxB, 0u, AGG);
  edge_kernel<32><<<EE / 128, 64, 0, stream>>>(xb, eattr, We0, be0, WpreP0, bpre0,
                                               src, dst, sumB, sqB, minB, maxB);
  node_kernel<32><<<NN / 16, 32, 0, stream>>>(x, sumB, sqB, minB, maxB, cnt,
                                              WpostP0, bpost0, WlinP0, blin0,
                                              h0, h0b);
  // ---- Layer 1 (F=64) ----
  fill_kernel<<<2048, 256, 0, stream>>>((unsigned*)sumB, 0u, AGG);
  fill_kernel<<<2048, 256, 0, stream>>>((unsigned*)sqB, 0u, AGG);
  fill_kernel<<<2048, 256, 0, stream>>>(minB, 0xFFFFFFFFu, AGG);
  fill_kernel<<<2048, 256, 0, stream>>>(maxB, 0u, AGG);
  edge_kernel<64><<<EE / 128, 64, 0, stream>>>(h0b, eattr, We1, be1, WpreP1, bpre1,
                                               src, dst, sumB, sqB, minB, maxB);
  node_kernel<64><<<NN / 16, 32, 0, stream>>>(h0, sumB, sqB, minB, maxB, cnt,
                                              WpostP1, bpost1, WlinP1, blin1,
                                              h1, h1b);
  // ---- Pool + MLP ----
  pool_kernel<<<(NN * 64) / 256, 256, 0, stream>>>(h1, batch, gbuf);
  mlp_kernel<<<1, 64, 0, stream>>>(gbuf, hls, W1, b1, W2, b2, W3, b3, (float*)d_out);
}
